// UtilLayer_88441966559673
// MI455X (gfx1250) — matrix operand order repaired and checked by reading, not compiled, not run on hardware
//
#include <hip/hip_runtime.h>
#include <hip/hip_bf16.h>

typedef _Float16 v16h __attribute__((ext_vector_type(16)));
typedef _Float16 v8h  __attribute__((ext_vector_type(8)));
typedef float    v8f  __attribute__((ext_vector_type(8)));

#define WAVES 8
#define TPB (WAVES * 32)

// Convert two 8-float chunks (K=kh..kh+7 and K=16+kh..16+kh+7) into one f16
// A-fragment register block, matching the ISA 16-bit A 16x32 layout.
__device__ __forceinline__ v16h cvt8x2(const float* __restrict__ p0,
                                       const float* __restrict__ p1) {
  float4 f0 = *(const float4*)(p0);
  float4 f1 = *(const float4*)(p0 + 4);
  float4 f2 = *(const float4*)(p1);
  float4 f3 = *(const float4*)(p1 + 4);
  v16h r;
  r[0]  = (_Float16)f0.x; r[1]  = (_Float16)f0.y; r[2]  = (_Float16)f0.z; r[3]  = (_Float16)f0.w;
  r[4]  = (_Float16)f1.x; r[5]  = (_Float16)f1.y; r[6]  = (_Float16)f1.z; r[7]  = (_Float16)f1.w;
  r[8]  = (_Float16)f2.x; r[9]  = (_Float16)f2.y; r[10] = (_Float16)f2.z; r[11] = (_Float16)f2.w;
  r[12] = (_Float16)f3.x; r[13] = (_Float16)f3.y; r[14] = (_Float16)f3.z; r[15] = (_Float16)f3.w;
  return r;
}

__global__ void zero_q_kernel(float* __restrict__ q, int n) {
  int i = blockIdx.x * blockDim.x + threadIdx.x;
  if (i < n) q[i] = 0.0f;
}

// ---------------- Edge MLP kernel ----------------
// Rows r in [0, 2E): r = 2*e + variant (variant 0 = edge_feats_u, 1 = reflected).
// Per row: h1 = relu(x@W1 + b1)  [192->128], h2 = relu(h1@W3 + b3) [128->128],
// out = h2 . W2[:,col] + b2[col] with col = (variant ? ad*10+as : as*10+ad).
// Contribution: q[graph[dst]] += 0.25 * out.
__global__ __launch_bounds__(TPB) void edge_kernel(
    const float* __restrict__ x, const float* __restrict__ xr,
    const int* __restrict__ src, const int* __restrict__ dst,
    const int* __restrict__ ja, const int* __restrict__ gid,
    const float* __restrict__ w1, const float* __restrict__ b1,
    const float* __restrict__ w3, const float* __restrict__ b3,
    const float* __restrict__ w2, const float* __restrict__ b2,
    float* __restrict__ q, int E, int iters) {
  __shared__ __align__(32) _Float16 w1f[6][8][32][16];   // 48 KB, B-fragment order
  __shared__ __align__(32) _Float16 w3f[4][8][32][16];   // 32 KB
  __shared__ __align__(32) _Float16 w2t[100][128];       // 25.6 KB, transposed
  __shared__ __align__(32) _Float16 hst[WAVES][16][128]; // 32 KB staging
  __shared__ float b1s[128];
  __shared__ float b3s[128];
  __shared__ float b2s[100];

  const int tid = threadIdx.x;
  // Stage weights into LDS in B-fragment layout: lane = n%16 + 16*((k>>3)&1),
  // elem = (k&7) + 8*((k>>4)&1), per ISA 16-bit B layout for 16x16x32.
  for (int f = tid; f < 192 * 128; f += TPB) {
    int k = f >> 7, n = f & 127;
    w1f[k >> 5][n >> 4][(n & 15) + ((k & 8) ? 16 : 0)][(k & 7) + ((k & 16) ? 8 : 0)] = (_Float16)w1[f];
  }
  for (int f = tid; f < 128 * 128; f += TPB) {
    int k = f >> 7, n = f & 127;
    w3f[k >> 5][n >> 4][(n & 15) + ((k & 8) ? 16 : 0)][(k & 7) + ((k & 16) ? 8 : 0)] = (_Float16)w3[f];
  }
  for (int f = tid; f < 100 * 128; f += TPB) {
    int c = f >> 7, k = f & 127;
    w2t[c][k] = (_Float16)w2[k * 100 + c];
  }
  if (tid < 128) { b1s[tid] = b1[tid]; b3s[tid] = b3[tid]; }
  if (tid < 100) { b2s[tid] = b2[tid]; }
  __syncthreads();

  const int w = tid >> 5;
  const int lane = tid & 31;
  const int m = lane & 15;                 // row within 16-row tile
  const int kh = (lane < 16) ? 0 : 8;      // A-fragment K-half offset
  const long totalRows = 2L * (long)E;

  for (int it = 0; it < iters; ++it) {
    const long tile = (long)blockIdx.x * (long)(WAVES * iters) + (long)it * WAVES + w;
    const long r0 = tile * 16;
    if (r0 + 16 > totalRows) break;

    // Prefetch next iteration's row data (global_prefetch_b8).
    {
      const long pr0 = (tile + WAVES) * 16;
      if (pr0 + 16 <= totalRows) {
        const long pr = pr0 + m;
        const float* pp = (((pr & 1) ? xr : x) + (pr >> 1) * 192);
        __builtin_prefetch(pp + kh, 0, 0);
        __builtin_prefetch(pp + 96 + kh, 0, 0);
      }
    }

    const long r = r0 + m;
    const float* xrow = ((r & 1) ? xr : x) + (r >> 1) * 192;

    // A fragments for GEMM1: K = 192 -> 6 tiles of 32.
    v16h a1[6];
#pragma unroll
    for (int kt = 0; kt < 6; ++kt)
      a1[kt] = cvt8x2(xrow + kt * 32 + kh, xrow + kt * 32 + 16 + kh);

    // GEMM1: [16x192] x [192x128] + bias, ReLU -> hst (f16).
#pragma unroll
    for (int nt = 0; nt < 8; ++nt) {
      v8f acc = {0.f, 0.f, 0.f, 0.f, 0.f, 0.f, 0.f, 0.f};
#pragma unroll
      for (int kt = 0; kt < 6; ++kt) {
        v16h bfr = *(const v16h*)(&w1f[kt][nt][lane][0]);
        acc = __builtin_amdgcn_wmma_f32_16x16x32_f16(false, a1[kt], false, bfr,
                                                     (short)0, acc, false, false);
      }
      const float bias = b1s[nt * 16 + m];
      const int rb = (lane >> 4) * 8;  // C layout: row = v + 8*(lane/16)
#pragma unroll
      for (int v = 0; v < 8; ++v) {
        float h = acc[v] + bias;
        hst[w][rb + v][nt * 16 + m] = (_Float16)(h > 0.f ? h : 0.f);
      }
    }
    asm volatile("s_wait_dscnt 0" ::: "memory");

    // Reload h1 as A fragments: K = 128 -> 4 tiles of 32.
    v16h a2[4];
#pragma unroll
    for (int kt = 0; kt < 4; ++kt) {
      v8h lo = *(const v8h*)(&hst[w][m][kt * 32 + kh]);
      v8h hi = *(const v8h*)(&hst[w][m][kt * 32 + 16 + kh]);
      v16h t;
#pragma unroll
      for (int i = 0; i < 8; ++i) { t[i] = lo[i]; t[i + 8] = hi[i]; }
      a2[kt] = t;
    }

    // GEMM2: [16x128] x [128x128] + bias, ReLU -> hst (overwrite).
#pragma unroll
    for (int nt = 0; nt < 8; ++nt) {
      v8f acc = {0.f, 0.f, 0.f, 0.f, 0.f, 0.f, 0.f, 0.f};
#pragma unroll
      for (int kt = 0; kt < 4; ++kt) {
        v16h bfr = *(const v16h*)(&w3f[kt][nt][lane][0]);
        acc = __builtin_amdgcn_wmma_f32_16x16x32_f16(false, a2[kt], false, bfr,
                                                     (short)0, acc, false, false);
      }
      const float bias = b3s[nt * 16 + m];
      const int rb = (lane >> 4) * 8;
#pragma unroll
      for (int v = 0; v < 8; ++v) {
        float h = acc[v] + bias;
        hst[w][rb + v][nt * 16 + m] = (_Float16)(h > 0.f ? h : 0.f);
      }
    }
    asm volatile("s_wait_dscnt 0" ::: "memory");

    // Final layer: only one output column per row. Lanes l and l+16 split the
    // 128-length dot, combine with a wave32 xor-shuffle, one atomic per row.
    {
      const long ee = r >> 1;
      const int variant = (int)(r & 1);
      const int s = src[ee], d = dst[ee];
      const int as = ja[s], ad = ja[d];
      const int g = gid[d];
      const int col = variant ? (ad * 10 + as) : (as * 10 + ad);
      const int kb = (lane >> 4) * 64;
      float partial = 0.f;
#pragma unroll
      for (int k = 0; k < 64; k += 8) {
        v8h hv = *(const v8h*)(&hst[w][m][kb + k]);
        v8h wv = *(const v8h*)(&w2t[col][kb + k]);
#pragma unroll
        for (int i = 0; i < 8; ++i) partial += (float)hv[i] * (float)wv[i];
      }
      const float tot = partial + __shfl_xor(partial, 16, 32);
      if (lane < 16) atomicAdd(&q[g], 0.25f * (tot + b2s[col]));
    }
  }
}

// ---------------- Node MLP kernel ----------------
// Per node n: h1 = relu(nf@W1+b1) [128->128], h2 = relu(h1@W3+b3),
// out = h2 . W2[:, ja[n]] + b2[ja[n]];  q[graph[n]] += out.
__global__ __launch_bounds__(TPB) void node_kernel(
    const float* __restrict__ nf,
    const int* __restrict__ ja, const int* __restrict__ gid,
    const float* __restrict__ w1, const float* __restrict__ b1,
    const float* __restrict__ w3, const float* __restrict__ b3,
    const float* __restrict__ w2, const float* __restrict__ b2,
    float* __restrict__ q, int N, int iters) {
  __shared__ __align__(32) _Float16 w1f[4][8][32][16];   // 32 KB
  __shared__ __align__(32) _Float16 w3f[4][8][32][16];   // 32 KB
  __shared__ __align__(32) _Float16 w2t[10][128];        // 2.5 KB
  __shared__ __align__(32) _Float16 hst[WAVES][16][128]; // 32 KB
  __shared__ float b1s[128];
  __shared__ float b3s[128];
  __shared__ float b2s[10];

  const int tid = threadIdx.x;
  for (int f = tid; f < 128 * 128; f += TPB) {
    int k = f >> 7, n = f & 127;
    w1f[k >> 5][n >> 4][(n & 15) + ((k & 8) ? 16 : 0)][(k & 7) + ((k & 16) ? 8 : 0)] = (_Float16)w1[f];
    w3f[k >> 5][n >> 4][(n & 15) + ((k & 8) ? 16 : 0)][(k & 7) + ((k & 16) ? 8 : 0)] = (_Float16)w3[f];
  }
  for (int f = tid; f < 10 * 128; f += TPB) {
    int c = f >> 7, k = f & 127;
    w2t[c][k] = (_Float16)w2[k * 10 + c];
  }
  if (tid < 128) { b1s[tid] = b1[tid]; b3s[tid] = b3[tid]; }
  if (tid < 10) { b2s[tid] = b2[tid]; }
  __syncthreads();

  const int w = tid >> 5;
  const int lane = tid & 31;
  const int m = lane & 15;
  const int kh = (lane < 16) ? 0 : 8;

  for (int it = 0; it < iters; ++it) {
    const long tile = (long)blockIdx.x * (long)(WAVES * iters) + (long)it * WAVES + w;
    const long r0 = tile * 16;
    if (r0 + 16 > (long)N) break;

    const long n = r0 + m;
    const float* xrow = nf + n * 128;

    v16h a1[4];
#pragma unroll
    for (int kt = 0; kt < 4; ++kt)
      a1[kt] = cvt8x2(xrow + kt * 32 + kh, xrow + kt * 32 + 16 + kh);

#pragma unroll
    for (int nt = 0; nt < 8; ++nt) {
      v8f acc = {0.f, 0.f, 0.f, 0.f, 0.f, 0.f, 0.f, 0.f};
#pragma unroll
      for (int kt = 0; kt < 4; ++kt) {
        v16h bfr = *(const v16h*)(&w1f[kt][nt][lane][0]);
        acc = __builtin_amdgcn_wmma_f32_16x16x32_f16(false, a1[kt], false, bfr,
                                                     (short)0, acc, false, false);
      }
      const float bias = b1s[nt * 16 + m];
      const int rb = (lane >> 4) * 8;
#pragma unroll
      for (int v = 0; v < 8; ++v) {
        float h = acc[v] + bias;
        hst[w][rb + v][nt * 16 + m] = (_Float16)(h > 0.f ? h : 0.f);
      }
    }
    asm volatile("s_wait_dscnt 0" ::: "memory");

    v16h a2[4];
#pragma unroll
    for (int kt = 0; kt < 4; ++kt) {
      v8h lo = *(const v8h*)(&hst[w][m][kt * 32 + kh]);
      v8h hi = *(const v8h*)(&hst[w][m][kt * 32 + 16 + kh]);
      v16h t;
#pragma unroll
      for (int i = 0; i < 8; ++i) { t[i] = lo[i]; t[i + 8] = hi[i]; }
      a2[kt] = t;
    }

#pragma unroll
    for (int nt = 0; nt < 8; ++nt) {
      v8f acc = {0.f, 0.f, 0.f, 0.f, 0.f, 0.f, 0.f, 0.f};
#pragma unroll
      for (int kt = 0; kt < 4; ++kt) {
        v16h bfr = *(const v16h*)(&w3f[kt][nt][lane][0]);
        acc = __builtin_amdgcn_wmma_f32_16x16x32_f16(false, a2[kt], false, bfr,
                                                     (short)0, acc, false, false);
      }
      const float bias = b3s[nt * 16 + m];
      const int rb = (lane >> 4) * 8;
#pragma unroll
      for (int v = 0; v < 8; ++v) {
        float h = acc[v] + bias;
        hst[w][rb + v][nt * 16 + m] = (_Float16)(h > 0.f ? h : 0.f);
      }
    }
    asm volatile("s_wait_dscnt 0" ::: "memory");

    {
      const int col = ja[n];
      const int g = gid[n];
      const int kb = (lane >> 4) * 64;
      float partial = 0.f;
#pragma unroll
      for (int k = 0; k < 64; k += 8) {
        v8h hv = *(const v8h*)(&hst[w][m][kb + k]);
        v8h wv = *(const v8h*)(&w2t[col][kb + k]);
#pragma unroll
        for (int i = 0; i < 8; ++i) partial += (float)hv[i] * (float)wv[i];
      }
      const float tot = partial + __shfl_xor(partial, 16, 32);
      if (lane < 16) atomicAdd(&q[g], tot + b2s[col]);
    }
  }
}

extern "C" void kernel_launch(void* const* d_in, const int* in_sizes, int n_in,
                              void* d_out, int out_size, void* d_ws, size_t ws_size,
                              hipStream_t stream) {
  const float* x   = (const float*)d_in[0];   // edge_feats_u        [E, 192]
  const float* xr  = (const float*)d_in[1];   // edge_feat_reflected [E, 192]
  const float* nf  = (const float*)d_in[2];   // node_feats_u        [N, 128]
  const int*   src = (const int*)d_in[3];
  const int*   dst = (const int*)d_in[4];
  const int*   ja  = (const int*)d_in[5];
  const int*   gid = (const int*)d_in[6];
  // d_in[7] = num_graphs (scalar) — output size already gives B.
  const float* ju1w = (const float*)d_in[8];
  const float* ju1b = (const float*)d_in[9];
  const float* ju3w = (const float*)d_in[10];
  const float* ju3b = (const float*)d_in[11];
  const float* ju2w = (const float*)d_in[12];
  const float* ju2b = (const float*)d_in[13];
  const float* iu1w = (const float*)d_in[14];
  const float* iu1b = (const float*)d_in[15];
  const float* iu3w = (const float*)d_in[16];
  const float* iu3b = (const float*)d_in[17];
  const float* iu2w = (const float*)d_in[18];
  const float* iu2b = (const float*)d_in[19];

  const int E = in_sizes[0] / 192;
  const int N = in_sizes[2] / 128;
  const int B = out_size;
  float* q = (float*)d_out;

  zero_q_kernel<<<(B + 255) / 256, 256, 0, stream>>>(q, B);

  const int ITERS = 4;
  const long tilesE = (2L * (long)E + 15) / 16;
  const int blocksE = (int)((tilesE + (long)(WAVES * ITERS) - 1) / (long)(WAVES * ITERS));
  edge_kernel<<<blocksE, TPB, 0, stream>>>(x, xr, src, dst, ja, gid,
                                           ju1w, ju1b, ju3w, ju3b, ju2w, ju2b,
                                           q, E, ITERS);

  const long tilesN = ((long)N + 15) / 16;
  const int blocksN = (int)((tilesN + (long)(WAVES * ITERS) - 1) / (long)(WAVES * ITERS));
  node_kernel<<<blocksN, TPB, 0, stream>>>(nf, ja, gid,
                                           iu1w, iu1b, iu3w, iu3b, iu2w, iu2b,
                                           q, N, ITERS);
}
